// GlobalBlockViT_14276471292712
// MI455X (gfx1250) — matrix operand order repaired and checked by reading, not compiled, not run on hardware
//
#include <hip/hip_runtime.h>

// B=16, N=4096, C=256, HEADS=8, HD=32, WS=8, NQ=8, POOL=4
// windows: 1024 total, 64 tokens each, kv_len = 72 (pad kv->80 logits / 96 AV-K)

#define DEVINL __device__ __forceinline__

typedef __attribute__((ext_vector_type(16))) _Float16 v16h;
typedef __attribute__((ext_vector_type(8)))  float    v8f;

DEVINL int lane_id() { return threadIdx.x & 31; }

DEVINL float wred(float v) {
#pragma unroll
  for (int o = 16; o >= 1; o >>= 1) v += __shfl_xor(v, o, 32);
  return v;
}

// ---- A-fragment (16x32 f16, MxK): documented interleaved layout ----
// lane: m = lane&15, hi = lane>>4; elem e<8 -> K = hi*8+e ; e>=8 -> K = 16+hi*8+(e-8)
DEVINL v16h fragA_h(const _Float16* base, int ld) {
  int l = lane_id(); int m = l & 15, hi = l >> 4;
  const _Float16* p = base + m * ld + hi * 8;
  v16h f;
#pragma unroll
  for (int e = 0; e < 8; ++e) f[e] = p[e];
#pragma unroll
  for (int e = 0; e < 8; ++e) f[e + 8] = p[e + 16];
  return f;
}

DEVINL v16h fragA_f(const float* base, int ld) {
  int l = lane_id(); int m = l & 15, hi = l >> 4;
  const float* p = base + m * ld + hi * 8;
  v16h f;
#pragma unroll
  for (int e = 0; e < 8; ++e) f[e] = (_Float16)p[e];
#pragma unroll
  for (int e = 0; e < 8; ++e) f[e + 8] = (_Float16)p[e + 16];
  return f;
}

// ---- B-fragment (32x16 f16, KxN): column n contiguous along K in source ----
// lane: n = lane&15, hi = lane>>4; elem e<8 -> K = hi*8+e ; e>=8 -> K = 16+hi*8+(e-8)
DEVINL v16h fragB_h(const _Float16* base, int ld) {
  int l = lane_id(); int n = l & 15, hi = l >> 4;
  const _Float16* p = base + n * ld + hi * 8;
  v16h f;
#pragma unroll
  for (int e = 0; e < 8; ++e) f[e] = p[e];
#pragma unroll
  for (int e = 0; e < 8; ++e) f[e + 8] = p[e + 16];
  return f;
}

DEVINL v8f wmma(v16h a, v16h b, v8f c) {
  return __builtin_amdgcn_wmma_f32_16x16x32_f16(false, a, false, b, (short)0, c,
                                                false, false);
}

// ---------------- kernel 0: fp32 -> f16 weight convert ----------------
__global__ void k_f2h(const float* __restrict__ s, _Float16* __restrict__ d, int n) {
  int i = blockIdx.x * 256 + threadIdx.x;
  if (i < n) d[i] = (_Float16)s[i];
}

// ---------------- kernel 1: LayerNorm -> f16 output (wave per token) ----------------
__global__ void __launch_bounds__(256) k_ln(const float* __restrict__ x,
                                            const float* __restrict__ g,
                                            const float* __restrict__ bta,
                                            _Float16* __restrict__ y) {
  int wave = threadIdx.x >> 5, l = lane_id();
  long t = (long)blockIdx.x * 8 + wave;
  const float* px = x + t * 256;
  float v[8]; float s = 0.f;
#pragma unroll
  for (int k = 0; k < 8; ++k) { v[k] = px[l + 32 * k]; s += v[k]; }
  s = wred(s) * (1.f / 256.f);
  float q = 0.f;
#pragma unroll
  for (int k = 0; k < 8; ++k) { float d = v[k] - s; q += d * d; }
  q = wred(q) * (1.f / 256.f);
  float inv = rsqrtf(q + 1e-5f);
  _Float16* py = y + t * 256;
#pragma unroll
  for (int k = 0; k < 8; ++k) {
    int c = l + 32 * k;
    py[c] = (_Float16)((v[k] - s) * inv * g[c] + bta[c]);
  }
}

// ---------------- kernel 2: 4x4 average pool (wave per pooled token) ----------------
__global__ void __launch_bounds__(256) k_pool(const _Float16* __restrict__ xnh,
                                              float* __restrict__ pooled) {
  int wave = threadIdx.x >> 5, l = lane_id();
  int idx = blockIdx.x * 8 + wave;  // b*256 + pt
  int b = idx >> 8, pt = idx & 255;
  int py = pt >> 4, px = pt & 15;
  const _Float16* base = xnh + ((long)b * 4096) * 256;
  float acc[8] = {0.f, 0.f, 0.f, 0.f, 0.f, 0.f, 0.f, 0.f};
  for (int i = 0; i < 4; ++i)
    for (int j = 0; j < 4; ++j) {
      int n = (py * 4 + i) * 64 + (px * 4 + j);
      const _Float16* p = base + (long)n * 256;
#pragma unroll
      for (int k = 0; k < 8; ++k) acc[k] += (float)p[l + 32 * k];
    }
  float* o = pooled + (long)idx * 256;
#pragma unroll
  for (int k = 0; k < 8; ++k) o[l + 32 * k] = acc[k] * (1.f / 16.f);
}

// ---------------- kernel 3: pooled k/v projection + folded BN (WMMA) ----------------
__global__ void __launch_bounds__(256) k_gemm_bn(
    const float* __restrict__ A, const _Float16* __restrict__ W,
    const float* __restrict__ pb, const float* __restrict__ g,
    const float* __restrict__ bt, const float* __restrict__ m,
    const float* __restrict__ vv, float* __restrict__ Y) {
  int wave = threadIdx.x >> 5;
  int gid = blockIdx.x * 8 + wave;          // b*256 + tile
  int b = gid >> 8, tile = gid & 255;
  int mt = tile >> 4, nt = tile & 15;
  const float* Ab = A + ((long)b * 256 + mt * 16) * 256;
  const _Float16* Wb = W + (long)nt * 16 * 256;
  v8f acc = {};
#pragma unroll
  for (int k0 = 0; k0 < 256; k0 += 32)
    acc = wmma(fragA_f(Ab + k0, 256), fragB_h(Wb + k0, 256), acc);
  int l = lane_id(); int n = l & 15, hi = l >> 4;
  int o = nt * 16 + n;
  float s = g[o] * rsqrtf(vv[o] + 1e-5f);
  float t = (pb[o] - m[o]) * s + bt[o];
  float* Yb = Y + ((long)b * 256 + mt * 16) * 256 + o;
#pragma unroll
  for (int r = 0; r < 8; ++r) Yb[(r + 8 * hi) * 256] = acc[r] * s + t;
}

// ---------------- kernel 4: global cross-attention (tiny, fp32) ----------------
__global__ void __launch_bounds__(256) k_cross(const float* __restrict__ qt,
                                               const float* __restrict__ kp,
                                               const float* __restrict__ vp,
                                               float* __restrict__ sbuf) {
  __shared__ float sc[8 * 256];
  int b = blockIdx.x, h = blockIdx.y;
  int tid = threadIdx.x;
  const float* kb = kp + (long)b * 65536 + h * 32;
  const float* vb = vp + (long)b * 65536 + h * 32;
  const float SCALE = 0.17677669529663687f;  // 32^-0.5
  for (int e = tid; e < 2048; e += 256) {
    int qi = e >> 8, t = e & 255;
    const float* qrow = qt + qi * 256 + h * 32;
    const float* krow = kb + (long)t * 256;
    float s = 0.f;
#pragma unroll
    for (int d = 0; d < 32; ++d) s += qrow[d] * krow[d];
    sc[e] = s * SCALE;
  }
  __syncthreads();
  int wave = tid >> 5, l = tid & 31;
  {
    float mx = -1e30f;
    for (int t = l; t < 256; t += 32) mx = fmaxf(mx, sc[wave * 256 + t]);
#pragma unroll
    for (int o = 16; o >= 1; o >>= 1) mx = fmaxf(mx, __shfl_xor(mx, o, 32));
    float sm = 0.f;
    for (int t = l; t < 256; t += 32) {
      float e_ = __expf(sc[wave * 256 + t] - mx);
      sc[wave * 256 + t] = e_; sm += e_;
    }
    sm = wred(sm);
    float inv = 1.f / sm;
    for (int t = l; t < 256; t += 32) sc[wave * 256 + t] *= inv;
  }
  __syncthreads();
  int qi = tid >> 5, d = tid & 31;
  float o = 0.f;
  for (int t = 0; t < 256; ++t) o += sc[qi * 256 + t] * vb[(long)t * 256 + d];
  sbuf[(b * 8 + qi) * 256 + h * 32 + d] = o + qt[qi * 256 + h * 32 + d];
}

// ---------------- kernel 5: gk/gv linear -> f16 output (WMMA) ----------------
__global__ void __launch_bounds__(256) k_gemm_small(const float* __restrict__ A,
                                                    const _Float16* __restrict__ W,
                                                    const float* __restrict__ bias,
                                                    _Float16* __restrict__ Y) {
  int wave = threadIdx.x >> 5;
  int tile = blockIdx.x * 8 + wave;  // 0..127
  int mt = tile >> 4, nt = tile & 15;
  const float* Ab = A + (long)mt * 16 * 256;
  const _Float16* Wb = W + (long)nt * 16 * 256;
  v8f acc = {};
#pragma unroll
  for (int k0 = 0; k0 < 256; k0 += 32)
    acc = wmma(fragA_f(Ab + k0, 256), fragB_h(Wb + k0, 256), acc);
  int l = lane_id(); int n = l & 15, hi = l >> 4;
  int o = nt * 16 + n;
  float bb = bias[o];
#pragma unroll
  for (int r = 0; r < 8; ++r)
    Y[(mt * 16 + r + 8 * hi) * 256 + o] = (_Float16)(acc[r] + bb);
}

// ---------------- kernel 6: fused window attention (WMMA everywhere) ----------------
__global__ void __launch_bounds__(256) k_winattn(
    const float* __restrict__ x, const _Float16* __restrict__ xnh,
    const _Float16* __restrict__ qkvW, const float* __restrict__ qkvB,
    const _Float16* __restrict__ projW, const float* __restrict__ projB,
    const _Float16* __restrict__ gkh, const _Float16* __restrict__ gvh,
    const float* __restrict__ rpb, float* __restrict__ xout,
    float* __restrict__ attnOut) {
  __shared__ __align__(16) _Float16 xh[64 * 256];  // 32 KB window tokens (f16)
  __shared__ __align__(16) _Float16 qh[64 * 256];  // 32 KB q * scale
  __shared__ __align__(16) _Float16 kh[80 * 256];  // 40 KB kv rows (64 win + 8 glb + pad)
  __shared__ __align__(16) _Float16 vT[256 * 96];  // 48 KB v^T [c][kv], kv pad 96
  __shared__ __align__(16) float    lg[64 * 80];   // 20 KB logits/probs scratch
  __shared__ __align__(16) _Float16 pr[64 * 96];   // 12 KB probs f16, K-padded
  __shared__ __align__(16) _Float16 ob[64 * 256];  // 32 KB attention output f16

  int w = blockIdx.x;
  int b = w >> 6, wy = (w >> 3) & 7, wx = w & 7;
  int tid = threadIdx.x, wave = tid >> 5, l = tid & 31;
  const float SCALE = 0.17677669529663687f;

  // ---- stage A: gather window tokens (f16, vectorized) ----
  {
    const _Float16* base = xnh + ((long)b * 4096) * 256;
    for (int i = tid * 8; i < 64 * 256; i += 256 * 8) {
      int t = i >> 8, c = i & 255;
      int n = (wy * 8 + (t >> 3)) * 64 + wx * 8 + (t & 7);
      *(uint4*)(&xh[t * 256 + c]) = *(const uint4*)(base + (long)n * 256 + c);
    }
    // global kv rows 64..71 + zero pad rows 72..79
    const _Float16* gkb = gkh + (long)b * 8 * 256;
    const _Float16* gvb = gvh + (long)b * 8 * 256;
    {
      int i = tid * 8;  // 2048 halves = 256 x uint4
      *(uint4*)(&kh[64 * 256 + i]) = *(const uint4*)(gkb + i);
      uint4 z = {0u, 0u, 0u, 0u};
      *(uint4*)(&kh[72 * 256 + i]) = z;
    }
    for (int i = tid; i < 256 * 32; i += 256) {   // vT cols 64..95
      int c = i >> 5, kv = i & 31;
      vT[c * 96 + 64 + kv] = (kv < 8) ? gvb[kv * 256 + c] : (_Float16)0.f;
    }
  }
  __syncthreads();

  // ---- stage B: QKV GEMM 64x256 @ 768x256^T, 192 tiles ----
  for (int it = 0; it < 24; ++it) {
    int tile = wave * 24 + it;
    int mt = tile / 48, ot = tile % 48;
    const _Float16* Ab = xh + mt * 16 * 256;
    const _Float16* Wb = qkvW + (long)ot * 16 * 256;
    __builtin_prefetch(qkvW + (long)(((tile + 1) % 48) * 16) * 256 + (l << 4), 0, 1);
    v8f acc = {};
#pragma unroll
    for (int k0 = 0; k0 < 256; k0 += 32)
      acc = wmma(fragA_h(Ab + k0, 256), fragB_h(Wb + k0, 256), acc);
    int n = l & 15, hi = l >> 4;
    int o = ot * 16 + n;
    float bb = qkvB[o];
#pragma unroll
    for (int r = 0; r < 8; ++r) {
      int t = mt * 16 + r + 8 * hi;
      float v = acc[r] + bb;
      if (o < 256)      qh[t * 256 + o]         = (_Float16)(v * SCALE);
      else if (o < 512) kh[t * 256 + (o - 256)] = (_Float16)v;
      else              vT[(o - 512) * 96 + t]  = (_Float16)v;  // transposed store
    }
  }
  __syncthreads();

  // ---- stage D: per-head attention ----
  for (int h = 0; h < 8; ++h) {
    // logits 4x5 tiles (kv padded to 80), single K=32 WMMA per tile
    for (int id = wave; id < 20; id += 8) {
      int mt = id / 5, nt = id % 5;
      v8f acc = {};
      acc = wmma(fragA_h(qh + mt * 16 * 256 + h * 32, 256),
                 fragB_h(kh + nt * 16 * 256 + h * 32, 256), acc);
      int n = l & 15, hi = l >> 4;
      int kv = nt * 16 + n;
#pragma unroll
      for (int r = 0; r < 8; ++r) {
        int t = mt * 16 + r + 8 * hi;
        float v = acc[r];
        if (kv < 64) {
          int di = (t >> 3) - (kv >> 3) + 7, dj = (t & 7) - (kv & 7) + 7;
          v += rpb[(di * 15 + dj) * 8 + h];
        }
        lg[t * 80 + kv] = v;
      }
    }
    __syncthreads();
    // softmax over 72; probs left in lg (fp32) and pr (f16, K-padded to 96)
    if (tid < 64) {
      int t = tid;
      float mx = -1e30f;
      for (int kv = 0; kv < 72; ++kv) mx = fmaxf(mx, lg[t * 80 + kv]);
      float sm = 0.f;
      for (int kv = 0; kv < 72; ++kv) sm += __expf(lg[t * 80 + kv] - mx);
      float inv = 1.f / sm;
      for (int kv = 0; kv < 72; ++kv) {
        float p = __expf(lg[t * 80 + kv] - mx) * inv;
        lg[t * 80 + kv] = p;
        pr[t * 96 + kv] = (_Float16)p;
      }
      for (int kv = 72; kv < 96; ++kv) pr[t * 96 + kv] = (_Float16)0.f;
    }
    __syncthreads();
    // coalesced fp32 prob store to d_out (all 256 threads)
    {
      float* ao = attnOut + ((long)w * 8 + h) * 64 * 72;
      for (int i = tid; i < 64 * 72; i += 256) {
        int t = i / 72, kv = i - t * 72;
        ao[i] = lg[t * 80 + kv];
      }
    }
    // A @ V : 4x2 tiles, one per wave, K = 96 (3 WMMA)
    {
      int mt = wave >> 1, dt = wave & 1;
      v8f acc = {};
#pragma unroll
      for (int k0 = 0; k0 < 96; k0 += 32)
        acc = wmma(fragA_h(pr + mt * 16 * 96 + k0, 96),
                   fragB_h(vT + (h * 32 + dt * 16) * 96 + k0, 96), acc);
      int n = l & 15, hi = l >> 4;
#pragma unroll
      for (int r = 0; r < 8; ++r)
        ob[(mt * 16 + r + 8 * hi) * 256 + h * 32 + dt * 16 + n] = (_Float16)acc[r];
    }
    __syncthreads();
  }

  // ---- stage E: output projection + residual with original x ----
  for (int it = 0; it < 8; ++it) {
    int tile = wave * 8 + it;  // 0..63
    int mt = tile >> 4, nt = tile & 15;
    v8f acc = {};
#pragma unroll
    for (int k0 = 0; k0 < 256; k0 += 32)
      acc = wmma(fragA_h(ob + mt * 16 * 256 + k0, 256),
                 fragB_h(projW + (long)nt * 16 * 256 + k0, 256), acc);
    int n = l & 15, hi = l >> 4;
    int o = nt * 16 + n;
    float bb = projB[o];
#pragma unroll
    for (int r = 0; r < 8; ++r) {
      int t = mt * 16 + r + 8 * hi;
      int ng = (wy * 8 + (t >> 3)) * 64 + wx * 8 + (t & 7);
      long off = ((long)b * 4096 + ng) * 256 + o;
      xout[off] = x[off] + acc[r] + bb;
    }
  }
}

// ---------------- kernel 7: fused LN2 + MLP + residual (in-place) ----------------
__global__ void __launch_bounds__(256) k_mlp(float* __restrict__ xio,
                                             const float* __restrict__ g2,
                                             const float* __restrict__ b2,
                                             const _Float16* __restrict__ w1,
                                             const float* __restrict__ b1v,
                                             const _Float16* __restrict__ w2,
                                             const float* __restrict__ b2v) {
  __shared__ __align__(16) _Float16 a0[64 * 256];   // 32 KB  LN2 output (f16)
  __shared__ __align__(16) _Float16 hb[64 * 1024];  // 128 KB hidden (f16)
  int tid = threadIdx.x, wave = tid >> 5, l = tid & 31;
  long base = (long)blockIdx.x * 64 * 256;
  // LN2: wave handles 8 rows
  for (int rr = 0; rr < 8; ++rr) {
    int t = wave * 8 + rr;
    const float* px = xio + base + (long)t * 256;
    float v[8]; float s = 0.f;
#pragma unroll
    for (int k = 0; k < 8; ++k) { v[k] = px[l + 32 * k]; s += v[k]; }
    s = wred(s) * (1.f / 256.f);
    float q = 0.f;
#pragma unroll
    for (int k = 0; k < 8; ++k) { float d = v[k] - s; q += d * d; }
    q = wred(q) * (1.f / 256.f);
    float inv = rsqrtf(q + 1e-5f);
#pragma unroll
    for (int k = 0; k < 8; ++k) {
      int c = l + 32 * k;
      a0[t * 256 + c] = (_Float16)((v[k] - s) * inv * g2[c] + b2[c]);
    }
  }
  __syncthreads();
  // fc1 + exact GELU: 4x64 tiles, K=256
  for (int it = 0; it < 32; ++it) {
    int tile = wave * 32 + it;
    int mt = tile >> 6, nt = tile & 63;
    __builtin_prefetch(w1 + (long)(((tile + 1) & 63) * 16) * 256 + (l << 4), 0, 1);
    v8f acc = {};
#pragma unroll
    for (int k0 = 0; k0 < 256; k0 += 32)
      acc = wmma(fragA_h(a0 + mt * 16 * 256 + k0, 256),
                 fragB_h(w1 + (long)nt * 16 * 256 + k0, 256), acc);
    int n = l & 15, hi = l >> 4;
    int o = nt * 16 + n;
    float bb = b1v[o];
#pragma unroll
    for (int r = 0; r < 8; ++r) {
      float u = acc[r] + bb;
      float ge = 0.5f * u * (1.f + erff(u * 0.7071067811865476f));
      hb[(mt * 16 + r + 8 * hi) * 1024 + o] = (_Float16)ge;
    }
  }
  __syncthreads();
  // fc2: 4x16 tiles, K=1024; residual in place
  for (int it = 0; it < 8; ++it) {
    int tile = wave * 8 + it;
    int mt = tile >> 4, nt = tile & 15;
    v8f acc = {};
#pragma unroll
    for (int k0 = 0; k0 < 1024; k0 += 32)
      acc = wmma(fragA_h(hb + mt * 16 * 1024 + k0, 1024),
                 fragB_h(w2 + (long)nt * 16 * 1024 + k0, 1024), acc);
    int n = l & 15, hi = l >> 4;
    int o = nt * 16 + n;
    float bb = b2v[o];
#pragma unroll
    for (int r = 0; r < 8; ++r) {
      int t = mt * 16 + r + 8 * hi;
      long off = base + (long)t * 256 + o;
      xio[off] = xio[off] + acc[r] + bb;
    }
  }
}

// ---------------- host launch ----------------
extern "C" void kernel_launch(void* const* d_in, const int* in_sizes, int n_in,
                              void* d_out, int out_size, void* d_ws, size_t ws_size,
                              hipStream_t stream) {
  (void)in_sizes; (void)n_in; (void)out_size; (void)ws_size;
  const float* x      = (const float*)d_in[0];
  const float* qt     = (const float*)d_in[1];
  const float* kprojW = (const float*)d_in[2];
  const float* kprojB = (const float*)d_in[3];
  const float* kbnG = (const float*)d_in[4];
  const float* kbnB = (const float*)d_in[5];
  const float* kbnM = (const float*)d_in[6];
  const float* kbnV = (const float*)d_in[7];
  const float* vprojW = (const float*)d_in[8];
  const float* vprojB = (const float*)d_in[9];
  const float* vbnG = (const float*)d_in[10];
  const float* vbnB = (const float*)d_in[11];
  const float* vbnM = (const float*)d_in[12];
  const float* vbnV = (const float*)d_in[13];
  const float* ckW = (const float*)d_in[14];
  const float* ckB = (const float*)d_in[15];
  const float* cvW = (const float*)d_in[16];
  const float* cvB = (const float*)d_in[17];
  const float* rpb = (const float*)d_in[18];
  const float* qkvW = (const float*)d_in[19];
  const float* qkvB = (const float*)d_in[20];
  const float* projW = (const float*)d_in[21];
  const float* projB = (const float*)d_in[22];
  const float* n1g = (const float*)d_in[23];
  const float* n1b = (const float*)d_in[24];
  const float* n2g = (const float*)d_in[25];
  const float* n2b = (const float*)d_in[26];
  const float* fc1W = (const float*)d_in[27];
  const float* fc1B = (const float*)d_in[28];
  const float* fc2W = (const float*)d_in[29];
  const float* fc2B = (const float*)d_in[30];

  char* ws = (char*)d_ws;
  size_t off = 0;
  auto alloc = [&](size_t bytes) -> void* {
    void* p = ws + off;
    off += (bytes + 255) & ~(size_t)255;
    return p;
  };
  _Float16* xnh   = (_Float16*)alloc(16l * 4096 * 256 * 2);  // 32 MB LN1 out (f16)
  float* pooled   = (float*)alloc(16l * 256 * 256 * 4);      // 4 MB
  float* kpool    = (float*)alloc(16l * 256 * 256 * 4);
  float* vpool    = (float*)alloc(16l * 256 * 256 * 4);
  float* sbuf     = (float*)alloc(128l * 256 * 4);
  _Float16* gkh   = (_Float16*)alloc(128l * 256 * 2);
  _Float16* gvh   = (_Float16*)alloc(128l * 256 * 2);
  _Float16* qkvWh   = (_Float16*)alloc(768l * 256 * 2);
  _Float16* projWh  = (_Float16*)alloc(256l * 256 * 2);
  _Float16* fc1Wh   = (_Float16*)alloc(1024l * 256 * 2);
  _Float16* fc2Wh   = (_Float16*)alloc(256l * 1024 * 2);
  _Float16* ckWh    = (_Float16*)alloc(256l * 256 * 2);
  _Float16* cvWh    = (_Float16*)alloc(256l * 256 * 2);
  _Float16* kprojWh = (_Float16*)alloc(256l * 256 * 2);
  _Float16* vprojWh = (_Float16*)alloc(256l * 256 * 2);

  float* xout = (float*)d_out;
  float* attnOut = xout + 16l * 4096 * 256;

  // weight conversions to f16
  k_f2h<<<768, 256, 0, stream>>>(qkvW, qkvWh, 768 * 256);
  k_f2h<<<256, 256, 0, stream>>>(projW, projWh, 256 * 256);
  k_f2h<<<1024, 256, 0, stream>>>(fc1W, fc1Wh, 1024 * 256);
  k_f2h<<<1024, 256, 0, stream>>>(fc2W, fc2Wh, 256 * 1024);
  k_f2h<<<256, 256, 0, stream>>>(ckW, ckWh, 256 * 256);
  k_f2h<<<256, 256, 0, stream>>>(cvW, cvWh, 256 * 256);
  k_f2h<<<256, 256, 0, stream>>>(kprojW, kprojWh, 256 * 256);
  k_f2h<<<256, 256, 0, stream>>>(vprojW, vprojWh, 256 * 256);

  k_ln<<<8192, 256, 0, stream>>>(x, n1g, n1b, xnh);
  k_pool<<<512, 256, 0, stream>>>(xnh, pooled);
  k_gemm_bn<<<512, 256, 0, stream>>>(pooled, kprojWh, kprojB, kbnG, kbnB, kbnM, kbnV, kpool);
  k_gemm_bn<<<512, 256, 0, stream>>>(pooled, vprojWh, vprojB, vbnG, vbnB, vbnM, vbnV, vpool);
  k_cross<<<dim3(16, 8), 256, 0, stream>>>(qt, kpool, vpool, sbuf);
  k_gemm_small<<<16, 256, 0, stream>>>(sbuf, ckWh, ckB, gkh);
  k_gemm_small<<<16, 256, 0, stream>>>(sbuf, cvWh, cvB, gvh);
  k_winattn<<<1024, 256, 0, stream>>>(x, xnh, qkvWh, qkvB, projWh, projB, gkh, gvh,
                                      rpb, xout, attnOut);
  k_mlp<<<1024, 256, 0, stream>>>(xout, n2g, n2b, fc1Wh, fc1B, fc2Wh, fc2B);
}
